// SetAbstraction_79860621902603
// MI455X (gfx1250) — matrix operand order repaired and checked by reading, not compiled, not run on hardware
//
#include <hip/hip_runtime.h>
#include <hip/hip_bf16.h>

typedef __attribute__((ext_vector_type(16))) _Float16 v16h;
typedef __attribute__((ext_vector_type(8)))  _Float16 v8h;
typedef __attribute__((ext_vector_type(8)))  float    v8f;

#define BB      16
#define NN      2048
#define DD      64
#define NPOINT  512
#define NSAMPLE 32
#define SC      67
#define PER_B   (NSAMPLE*NPOINT)      /* 16384 */
#define MCOLS   (BB*PER_B)            /* 262144 */
#define RADIUS2 0.04f
#define EPS_BN  1e-5f

#define APITCH  72                    /* halves; 144B, 16B multiple */
#define BPITCH  72

/* ---------------- workspace layout (float units) ---------------- */
constexpr size_t SZ_E    = (size_t)134*MCOLS;           /* 35,127,296 */
constexpr size_t OFF_E   = 0;
constexpr size_t OFF_X1  = OFF_E + SZ_E;
constexpr size_t OFF_X2  = 0;                           /* reuse E (conv0 input dead) */
constexpr size_t OFF_X3  = OFF_X1;                      /* reuse X1 (dead) + beyond   */
constexpr size_t PH1     = OFF_X1 + (size_t)256*MCOLS;  /* 102,236,160 */
constexpr size_t OFF_XN   = PH1;
constexpr size_t OFF_SXB  = OFF_XN   + (size_t)BB*NN;
constexpr size_t OFF_SPB  = OFF_SXB  + (size_t)BB*NPOINT*3;
constexpr size_t OFF_MEAN = OFF_SPB  + (size_t)BB*NPOINT*DD;
constexpr size_t OFF_RSTD = OFF_MEAN + 512;
constexpr size_t OFF_PART = OFF_RSTD + 512;
constexpr size_t OFF_FLAG = OFF_PART + 64;              /* NN ints */
constexpr size_t OFF_H1   = OFF_FLAG + NN;
constexpr size_t OFF_IDX  = OFF_H1   + (size_t)BB*SC*NN;   /* BB*NN*NSAMPLE ints */
constexpr size_t OFF_S    = OFF_IDX  + (size_t)BB*NN*NSAMPLE;
constexpr size_t OFF_INNER= OFF_S    + (size_t)BB*NPOINT*NN;

/* output positions */
constexpr size_t OUT_NEWXYZ = 0;                          /* B*3*NPOINT = 24576 */
constexpr size_t OUT_NEWPTS = 24576;                      /* B*256*NPOINT       */
constexpr size_t OUT_LOSS   = OUT_NEWPTS + (size_t)BB*256*NPOINT;
constexpr size_t OUT_UNIQ   = OUT_LOSS + 1;

/* ---------------- helpers ---------------- */
__device__ inline float blockReduceSum(float v, float* sm) {
    int t = threadIdx.x;
    sm[t] = v; __syncthreads();
    for (int s = 128; s > 0; s >>= 1) { if (t < s) sm[t] += sm[t + s]; __syncthreads(); }
    float r = sm[0]; __syncthreads();
    return r;
}

__device__ inline v8h cvt8(float4 a, float4 b) {
    v8h r;
    r[0] = (_Float16)a.x; r[1] = (_Float16)a.y; r[2] = (_Float16)a.z; r[3] = (_Float16)a.w;
    r[4] = (_Float16)b.x; r[5] = (_Float16)b.y; r[6] = (_Float16)b.z; r[7] = (_Float16)b.w;
    return r;
}

/* one 16x16x32 WMMA sub-step; A LDS is [m][k] (pitch APITCH),
   B LDS is TRANSPOSED [n][k] (pitch BPITCH) so both fragments are
   contiguous 16B chunks -> ds_load_b128                                 */
__device__ inline void wmma_step(const _Float16* As, const _Float16* Bts,
                                 int rowOff, int colOff, int ksub, v8f& acc) {
    int lane = threadIdx.x & 31;
    int lh = lane >> 4, lm = lane & 15;
    const _Float16* ar = As  + (rowOff + lm) * APITCH + ksub;
    const _Float16* br = Bts + (colOff + lm) * BPITCH + ksub;
    v8h a0 = *(const v8h*)(ar + 8 * lh);
    v8h a1 = *(const v8h*)(ar + 16 + 8 * lh);
    v8h b0 = *(const v8h*)(br + 8 * lh);
    v8h b1 = *(const v8h*)(br + 16 + 8 * lh);
    v16h a = __builtin_shufflevector(a0, a1, 0,1,2,3,4,5,6,7,8,9,10,11,12,13,14,15);
    v16h b = __builtin_shufflevector(b0, b1, 0,1,2,3,4,5,6,7,8,9,10,11,12,13,14,15);
    acc = __builtin_amdgcn_wmma_f32_16x16x32_f16(false, a, false, b, (short)0, acc,
                                                 false, false);
}

/* ---------------- pointwise / reduction kernels ---------------- */
__global__ void xn_kernel(const float* __restrict__ xyz, float* __restrict__ xn) {
    int t = blockIdx.x * 256 + threadIdx.x;
    if (t >= BB * NN) return;
    int b = t / NN, n = t % NN;
    const float* xb = xyz + (long)b * 3 * NN;
    float x0 = xb[n], x1 = xb[NN + n], x2 = xb[2 * NN + n];
    xn[t] = x0 * x0 + x1 * x1 + x2 * x2;
}

__global__ void ballquery_kernel(const float* __restrict__ xyz, const float* __restrict__ xn,
                                 int* __restrict__ idx) {
    int t = blockIdx.x * 256 + threadIdx.x;
    if (t >= BB * NN) return;
    int b = t / NN, n = t % NN;
    const float* xb = xyz + (long)b * 3 * NN;
    float x0 = xb[n], x1 = xb[NN + n], x2 = xb[2 * NN + n];
    float xs = xn[t];
    int* out = idx + (long)t * NSAMPLE;
    int cnt = 0, first = -1;
    for (int m = 0; m < NN; ++m) {
        float d = xs + xn[b * NN + m]
                - 2.f * (x0 * xb[m] + x1 * xb[NN + m] + x2 * xb[2 * NN + m]);
        if (d <= RADIUS2) {
            if (first < 0) first = m;
            out[cnt++] = m;
            if (cnt >= NSAMPLE) break;
        }
    }
    if (first < 0) first = n;
    for (int k = cnt; k < NSAMPLE; ++k) out[k] = first;
}

__global__ void h1_kernel(const float* __restrict__ points, const float* __restrict__ xyz,
                          const float* __restrict__ w1, float* __restrict__ h1) {
    long t = (long)blockIdx.x * 256 + threadIdx.x;
    if (t >= (long)BB * SC * NN) return;
    int n = t % NN; int o = (t / NN) % SC; int b = t / ((long)NN * SC);
    const float* wr = w1 + o * SC;
    float s = 0.f;
#pragma unroll 8
    for (int c = 0; c < DD; ++c) s += wr[c] * points[((long)b * DD + c) * NN + n];
    for (int c = 0; c < 3; ++c)  s += wr[DD + c] * xyz[((long)b * 3 + c) * NN + n];
    h1[t] = s;   /* bias omitted: cancels under batch-norm */
}

__global__ void bnstats_bcn_kernel(const float* __restrict__ X, int C,
                                   float* __restrict__ mean, float* __restrict__ rstd) {
    __shared__ float sm[256];
    int c = blockIdx.x;
    float s1 = 0.f, s2 = 0.f;
    const int M = BB * NN;
    for (int i = threadIdx.x; i < M; i += 256) {
        int b = i / NN, n = i % NN;
        float v = X[((long)b * C + c) * NN + n];
        s1 += v; s2 += v * v;
    }
    s1 = blockReduceSum(s1, sm);
    s2 = blockReduceSum(s2, sm);
    if (threadIdx.x == 0) {
        float m = s1 / M;
        mean[c] = m;
        rstd[c] = rsqrtf(s2 / M - m * m + EPS_BN);
    }
}

__global__ void bn_apply_bcn_kernel(float* __restrict__ X, int C,
                                    const float* __restrict__ g, const float* __restrict__ bb,
                                    const float* __restrict__ mean, const float* __restrict__ rstd) {
    long t = (long)blockIdx.x * 256 + threadIdx.x;
    if (t >= (long)BB * C * NN) return;
    int c = (t / NN) % C;
    X[t] = g[c] * (X[t] - mean[c]) * rstd[c] + bb[c];
}

__global__ void h2_kernel(const float* __restrict__ h1, const float* __restrict__ w2,
                          float* __restrict__ S) {
    long t = (long)blockIdx.x * 256 + threadIdx.x;
    if (t >= (long)BB * NPOINT * NN) return;
    int n = t % NN; int s = (t / NN) % NPOINT; int b = t / ((long)NN * NPOINT);
    const float* wr = w2 + s * SC;
    const float* hb = h1 + (long)b * SC * NN + n;
    float acc = 0.f;
#pragma unroll 8
    for (int c = 0; c < SC; ++c) acc += wr[c] * hb[(long)c * NN];
    S[t] = acc;
}

__global__ void softmax_kernel(float* __restrict__ S,
                               const float* __restrict__ g, const float* __restrict__ bb,
                               const float* __restrict__ mean, const float* __restrict__ rstd) {
    __shared__ float sm[256];
    int row = blockIdx.x;             /* b*NPOINT + s */
    int s = row % NPOINT;
    float* x = S + (long)row * NN;
    float ga = g[s], be = bb[s], mu = mean[s], rs = rstd[s];
    float mx = -1e30f;
    for (int n = threadIdx.x; n < NN; n += 256)
        mx = fmaxf(mx, ga * (x[n] - mu) * rs + be);
    sm[threadIdx.x] = mx; __syncthreads();
    for (int st = 128; st > 0; st >>= 1) {
        if (threadIdx.x < st) sm[threadIdx.x] = fmaxf(sm[threadIdx.x], sm[threadIdx.x + st]);
        __syncthreads();
    }
    mx = sm[0]; __syncthreads();
    float sum = 0.f;
    for (int n = threadIdx.x; n < NN; n += 256) {
        float v = __expf(ga * (x[n] - mu) * rs + be - mx);
        x[n] = v; sum += v;
    }
    sum = blockReduceSum(sum, sm);
    float inv = 1.f / sum;
    for (int n = threadIdx.x; n < NN; n += 256) x[n] *= inv;
}

__global__ void cosloss_kernel(const float* __restrict__ inner, float* __restrict__ partial) {
    __shared__ float nrm[NPOINT];
    __shared__ float sm[256];
    int b = blockIdx.x;
    const float* I = inner + (long)b * NPOINT * NPOINT;
    for (int s = threadIdx.x; s < NPOINT; s += 256)
        nrm[s] = sqrtf(fmaxf(I[(long)s * NPOINT + s], 0.f));
    __syncthreads();
    float acc = 0.f;
    for (int e = threadIdx.x; e < NPOINT * NPOINT; e += 256) {
        int s = e >> 9, t2 = e & 511;
        if (s != t2) {
            float c = I[e] / (nrm[s] * nrm[t2] + 1e-10f);
            acc += c * c;
        }
    }
    acc = blockReduceSum(acc, sm);
    if (threadIdx.x == 0) partial[b] = sqrtf(acc);
}

__global__ void finalize_loss_kernel(const float* __restrict__ partial, float* __restrict__ out) {
    if (threadIdx.x == 0) {
        float s = 0.f;
        for (int b = 0; b < BB; ++b) s += partial[b];
        out[0] = s / (float)BB;
    }
}

__global__ void zero_flags_kernel(int* __restrict__ flags) {
    int t = blockIdx.x * 256 + threadIdx.x;
    if (t < NN) flags[t] = 0;
}

__global__ void argmax_flag_kernel(const float* __restrict__ S, int* __restrict__ flags) {
    __shared__ float bv[256];
    __shared__ int   bi[256];
    int s = blockIdx.x;                     /* batch 0 only */
    const float* x = S + (long)s * NN;
    float best = -1e30f; int bidx = 0;
    for (int n = threadIdx.x; n < NN; n += 256) {
        float v = x[n];
        if (v > best) { best = v; bidx = n; }
    }
    bv[threadIdx.x] = best; bi[threadIdx.x] = bidx; __syncthreads();
    for (int st = 128; st > 0; st >>= 1) {
        if (threadIdx.x < st) {
            float ov = bv[threadIdx.x + st]; int oi = bi[threadIdx.x + st];
            if (ov > bv[threadIdx.x] || (ov == bv[threadIdx.x] && oi < bi[threadIdx.x])) {
                bv[threadIdx.x] = ov; bi[threadIdx.x] = oi;
            }
        }
        __syncthreads();
    }
    if (threadIdx.x == 0) flags[bi[0]] = 1;
}

__global__ void count_kernel(const int* __restrict__ flags, float* __restrict__ out) {
    __shared__ float sm[256];
    float c = 0.f;
    for (int n = threadIdx.x; n < NN; n += 256) c += (float)flags[n];
    c = blockReduceSum(c, sm);
    if (threadIdx.x == 0) out[0] = c;
}

__global__ void newxyz_kernel(const float* __restrict__ SX, float* __restrict__ out) {
    int t = blockIdx.x * 256 + threadIdx.x;
    if (t >= BB * 3 * NPOINT) return;
    int s = t % NPOINT; int c = (t / NPOINT) % 3; int b = t / (3 * NPOINT);
    out[t] = SX[((long)b * NPOINT + s) * 3 + c];
}

__global__ void edge_base_kernel(const float* __restrict__ SP, const float* __restrict__ SX,
                                 float* __restrict__ E) {
    int m = blockIdx.x * 256 + threadIdx.x;
    if (m >= MCOLS) return;
    int b = m / PER_B; int s = (m % PER_B) % NPOINT;
    const float* sp = SP + ((long)b * NPOINT + s) * DD;
#pragma unroll 8
    for (int c = 0; c < DD; ++c) E[(long)c * MCOLS + m] = sp[c];
    const float* sx = SX + ((long)b * NPOINT + s) * 3;
    for (int c = 0; c < 3; ++c) E[(long)(128 + c) * MCOLS + m] = sx[c];
}

__global__ void bnstats_rows_kernel(const float* __restrict__ X,
                                    float* __restrict__ mean, float* __restrict__ rstd) {
    __shared__ float sm[256];
    int c = blockIdx.x;
    const float* x = X + (long)c * MCOLS;
    float s1 = 0.f, s2 = 0.f;
    for (int i = threadIdx.x; i < MCOLS; i += 256) { float v = x[i]; s1 += v; s2 += v * v; }
    s1 = blockReduceSum(s1, sm);
    s2 = blockReduceSum(s2, sm);
    if (threadIdx.x == 0) {
        float m = s1 / (float)MCOLS;
        mean[c] = m;
        rstd[c] = rsqrtf(s2 / (float)MCOLS - m * m + EPS_BN);
    }
}

__global__ void bn_apply_leaky_kernel(float* __restrict__ X, int C,
                                      const float* __restrict__ g, const float* __restrict__ bb,
                                      const float* __restrict__ mean, const float* __restrict__ rstd) {
    long t = (long)blockIdx.x * 256 + threadIdx.x;
    if (t >= (long)C * MCOLS) return;
    int c = t / MCOLS;
    float v = g[c] * (X[t] - mean[c]) * rstd[c] + bb[c];
    X[t] = (v < 0.f) ? 0.2f * v : v;
}

__global__ void maxk_kernel(const float* __restrict__ X3, float* __restrict__ out) {
    long t = (long)blockIdx.x * 256 + threadIdx.x;
    if (t >= (long)BB * 256 * NPOINT) return;
    int s = t % NPOINT; int o = (t / NPOINT) % 256; int b = t / ((long)NPOINT * 256);
    const float* x = X3 + (long)o * MCOLS + (long)b * PER_B + s;
    float mx = -1e30f;
#pragma unroll
    for (int k = 0; k < NSAMPLE; ++k) mx = fmaxf(mx, x[(long)k * NPOINT]);
    out[t] = mx;
}

/* ---------------- WMMA GEMM kernels ----------------
   Block = 256 threads (8 wave32), macro tile 32(M) x 64(N), K staged 64. */

/* C[b] = A[b](MxK) * Bt[b]^T ; Bt is (Ncols x K) row-major.
   A is always the select matrix here: M,K full -> unguarded vector loads. */
__global__ __launch_bounds__(256) void wmma_gemm_bt(
    const float* __restrict__ A, int lda, long abatch,
    const float* __restrict__ Bt, int ldbt, long bbatch,
    float* __restrict__ C, int ldc, long cbatch,
    int M, int Ncols, int K)
{
    __shared__ __attribute__((aligned(16))) _Float16 As[32 * APITCH];
    __shared__ __attribute__((aligned(16))) _Float16 Bts[64 * BPITCH];
    int b = blockIdx.z;
    const float* Ab = A + (long)b * abatch;
    const float* Bb = Bt + (long)b * bbatch;
    float* Cb = C + (long)b * cbatch;
    int tM = blockIdx.y * 32, tN = blockIdx.x * 64;
    int wid = threadIdx.x >> 5;
    int rowOff = (wid >> 2) * 16, colOff = (wid & 3) * 16;
    int t = threadIdx.x;
    v8f acc = {};
    for (int k0 = 0; k0 < K; k0 += 64) {
        /* A: 32m x 64k, one 8-k chunk per thread, vector global + vector LDS */
        {
            int m = t >> 3, kq = (t & 7) * 8;
            const float* src = Ab + (long)(tM + m) * lda + k0 + kq;
            float4 f0 = *(const float4*)(src);
            float4 f1 = *(const float4*)(src + 4);
            *(v8h*)(As + m * APITCH + kq) = cvt8(f0, f1);
        }
        /* B: 64n x 64k transposed into LDS [n][k]; rows clamped, branchless */
        for (int i = t; i < 512; i += 256) {
            int n = i >> 3, kq = (i & 7) * 8;
            int gn = tN + n;
            bool ok = gn < Ncols;
            const float* src = Bb + (long)(ok ? gn : 0) * ldbt + k0 + kq;
            float4 f0 = *(const float4*)(src);
            float4 f1 = *(const float4*)(src + 4);
            v8h pk = cvt8(f0, f1);
            if (!ok) pk = (v8h)(_Float16)0.f;
            *(v8h*)(Bts + n * BPITCH + kq) = pk;
        }
        __syncthreads();
        wmma_step(As, Bts, rowOff, colOff, 0,  acc);
        wmma_step(As, Bts, rowOff, colOff, 32, acc);
        __syncthreads();
    }
    int lane = threadIdx.x & 31; int lh = lane >> 4, lm = lane & 15;
#pragma unroll
    for (int r = 0; r < 8; ++r) {
        int gm = tM + rowOff + r + 8 * lh;
        int gn = tN + colOff + lm;
        if (gm < M && gn < Ncols) Cb[(long)gm * ldc + gn] = acc[r];
    }
}

/* C = A(MxK, row) * B(KxN, row) — 1x1 convs. Weights may be K-unaligned:
   scalar clamped loads for A; B guarded on K only. */
__global__ __launch_bounds__(256) void wmma_gemm_nn(
    const float* __restrict__ A, int lda,
    const float* __restrict__ Bm, long ldb,
    float* __restrict__ C, long ldc,
    int M, long Ncols, int K)
{
    __shared__ __attribute__((aligned(16))) _Float16 As[32 * APITCH];
    __shared__ __attribute__((aligned(16))) _Float16 Bts[64 * BPITCH];
    int tM = blockIdx.y * 32;
    long tN = (long)blockIdx.x * 64;
    int wid = threadIdx.x >> 5;
    int rowOff = (wid >> 2) * 16, colOff = (wid & 3) * 16;
    int t = threadIdx.x;
    v8f acc = {};
    for (int k0 = 0; k0 < K; k0 += 64) {
        {
            int m = t >> 3, kq = (t & 7) * 8;
            int gm = tM + m;
            v8h pk;
#pragma unroll
            for (int i = 0; i < 8; ++i) {
                int gk = k0 + kq + i;
                bool ok = (gm < M) && (gk < K);
                float v = A[(long)(ok ? gm : 0) * lda + (ok ? gk : 0)];
                pk[i] = (_Float16)(ok ? v : 0.f);
            }
            *(v8h*)(As + m * APITCH + kq) = pk;
        }
        for (int i = t; i < 512; i += 256) {
            int n = i >> 3, kq = (i & 7) * 8;
            long gn = tN + n;
            v8h pk;
#pragma unroll
            for (int j = 0; j < 8; ++j) {
                int gk = k0 + kq + j;
                bool ok = gk < K;
                float v = Bm[(long)(ok ? gk : 0) * ldb + gn];
                pk[j] = (_Float16)(ok ? v : 0.f);
            }
            *(v8h*)(Bts + n * BPITCH + kq) = pk;
        }
        __syncthreads();
        wmma_step(As, Bts, rowOff, colOff, 0,  acc);
        wmma_step(As, Bts, rowOff, colOff, 32, acc);
        __syncthreads();
    }
    int lane = threadIdx.x & 31; int lh = lane >> 4, lm = lane & 15;
#pragma unroll
    for (int r = 0; r < 8; ++r) {
        int gm = tM + rowOff + r + 8 * lh;
        long gn = tN + colOff + lm;
        if (gm < M && gn < Ncols) C[(long)gm * ldc + gn] = acc[r];
    }
}

/* sgp/sgx: select(512x2048) x gathered grouped features (fused grouping);
   epilogue writes edge channels (value - sampled center). */
__global__ __launch_bounds__(256) void wmma_gemm_gather_edge(
    const float* __restrict__ S, const int* __restrict__ idx,
    const float* __restrict__ src,      /* (B, Cdim, NN) channel-major */
    const float* __restrict__ sub,      /* (B, NPOINT, Cdim)           */
    float* __restrict__ E, int Cdim, int chBase, int widthJ)
{
    __shared__ __attribute__((aligned(16))) _Float16 As[32 * APITCH];
    __shared__ __attribute__((aligned(16))) _Float16 Bts[64 * BPITCH];
    int b = blockIdx.z;
    const float* Ab = S + (long)b * NPOINT * NN;
    int tM = blockIdx.y * 32, tN = blockIdx.x * 64;
    int wid = threadIdx.x >> 5;
    int rowOff = (wid >> 2) * 16, colOff = (wid & 3) * 16;
    int t = threadIdx.x;
    v8f acc = {};
    for (int k0 = 0; k0 < NN; k0 += 64) {
        {
            int m = t >> 3, kq = (t & 7) * 8;
            const float* srcA = Ab + (long)(tM + m) * NN + k0 + kq;
            float4 f0 = *(const float4*)(srcA);
            float4 f1 = *(const float4*)(srcA + 4);
            *(v8h*)(As + m * APITCH + kq) = cvt8(f0, f1);
        }
        for (int i = t; i < 512; i += 256) {
            int n = i >> 3, kq = (i & 7) * 8;
            int gj = tN + n;
            bool ok = gj < widthJ;
            int jc = ok ? gj : 0;
            int ks = jc / Cdim, ch = jc - ks * Cdim;
            const float* srow = src + ((long)b * Cdim + ch) * NN;
            const int* irow = idx + (long)b * NN * NSAMPLE + ks;
            v8h pk;
#pragma unroll
            for (int j = 0; j < 8; ++j) {
                int gk = k0 + kq + j;
                int nidx = irow[(long)gk * NSAMPLE];
                float v = srow[nidx];
                pk[j] = (_Float16)(ok ? v : 0.f);
            }
            *(v8h*)(Bts + n * BPITCH + kq) = pk;
        }
        __syncthreads();
        wmma_step(As, Bts, rowOff, colOff, 0,  acc);
        wmma_step(As, Bts, rowOff, colOff, 32, acc);
        __syncthreads();
    }
    int lane = threadIdx.x & 31; int lh = lane >> 4, lm = lane & 15;
#pragma unroll
    for (int r = 0; r < 8; ++r) {
        int s = tM + rowOff + r + 8 * lh;
        int j = tN + colOff + lm;
        if (s < NPOINT && j < widthJ) {
            int ks = j / Cdim, ch = j - ks * Cdim;
            float val = acc[r] - sub[((long)b * NPOINT + s) * Cdim + ch];
            E[(long)(chBase + ch) * MCOLS + (long)b * PER_B + (long)ks * NPOINT + s] = val;
        }
    }
}

/* ---------------- launcher ---------------- */
extern "C" void kernel_launch(void* const* d_in, const int* in_sizes, int n_in,
                              void* d_out, int out_size, void* d_ws, size_t ws_size,
                              hipStream_t stream) {
    (void)in_sizes; (void)n_in; (void)out_size; (void)ws_size;
    const float* xyz    = (const float*)d_in[0];
    const float* points = (const float*)d_in[1];
    const float* w1_w   = (const float*)d_in[2];
    const float* bn1_g  = (const float*)d_in[4];
    const float* bn1_b  = (const float*)d_in[5];
    const float* w2_w   = (const float*)d_in[6];
    const float* bn2_g  = (const float*)d_in[8];
    const float* bn2_b  = (const float*)d_in[9];
    const float* conv_w[3] = { (const float*)d_in[10], (const float*)d_in[14], (const float*)d_in[18] };
    const float* bng[3]    = { (const float*)d_in[12], (const float*)d_in[16], (const float*)d_in[20] };
    const float* bnb[3]    = { (const float*)d_in[13], (const float*)d_in[17], (const float*)d_in[21] };

    float* w = (float*)d_ws;
    float* E     = w + OFF_E;
    float* X1    = w + OFF_X1;
    float* X2    = w + OFF_X2;
    float* X3    = w + OFF_X3;
    float* xn    = w + OFF_XN;
    float* SX    = w + OFF_SXB;
    float* SP    = w + OFF_SPB;
    float* meanb = w + OFF_MEAN;
    float* rstdb = w + OFF_RSTD;
    float* part  = w + OFF_PART;
    int*   flags = (int*)(w + OFF_FLAG);
    float* h1    = w + OFF_H1;
    int*   idx   = (int*)(w + OFF_IDX);
    float* S     = w + OFF_S;
    float* inner = w + OFF_INNER;
    float* out   = (float*)d_out;

    /* 1. ball query */
    xn_kernel<<<BB * NN / 256, 256, 0, stream>>>(xyz, xn);
    ballquery_kernel<<<BB * NN / 256, 256, 0, stream>>>(xyz, xn, idx);

    /* 2. h1 = w1 * [points;xyz], bn1 */
    h1_kernel<<<(int)(((long)BB * SC * NN + 255) / 256), 256, 0, stream>>>(points, xyz, w1_w, h1);
    bnstats_bcn_kernel<<<SC, 256, 0, stream>>>(h1, SC, meanb, rstdb);
    bn_apply_bcn_kernel<<<(int)(((long)BB * SC * NN + 255) / 256), 256, 0, stream>>>(
        h1, SC, bn1_g, bn1_b, meanb, rstdb);

    /* 3. h2 = w2 * h1, bn2 + softmax -> select (S) */
    h2_kernel<<<(int)((long)BB * NPOINT * NN / 256), 256, 0, stream>>>(h1, w2_w, S);
    bnstats_bcn_kernel<<<NPOINT, 256, 0, stream>>>(S, NPOINT, meanb, rstdb);
    softmax_kernel<<<BB * NPOINT, 256, 0, stream>>>(S, bn2_g, bn2_b, meanb, rstdb);

    /* 4. inner = S * S^T  (WMMA), cos_loss */
    {
        dim3 g(NPOINT / 64, NPOINT / 32, BB);
        wmma_gemm_bt<<<g, 256, 0, stream>>>(S, NN, (long)NPOINT * NN,
                                            S, NN, (long)NPOINT * NN,
                                            inner, NPOINT, (long)NPOINT * NPOINT,
                                            NPOINT, NPOINT, NN);
    }
    cosloss_kernel<<<BB, 256, 0, stream>>>(inner, part);
    finalize_loss_kernel<<<1, 32, 0, stream>>>(part, out + OUT_LOSS);

    /* 5. unique_num (batch 0 argmax) */
    zero_flags_kernel<<<NN / 256, 256, 0, stream>>>(flags);
    argmax_flag_kernel<<<NPOINT, 256, 0, stream>>>(S, flags);
    count_kernel<<<1, 256, 0, stream>>>(flags, out + OUT_UNIQ);

    /* 6. sample_points / sample_xyz (WMMA, B transposed from channel-major inputs) */
    {
        dim3 g(1, NPOINT / 32, BB);
        wmma_gemm_bt<<<g, 256, 0, stream>>>(S, NN, (long)NPOINT * NN,
                                            points, NN, (long)DD * NN,
                                            SP, DD, (long)NPOINT * DD,
                                            NPOINT, DD, NN);
        wmma_gemm_bt<<<g, 256, 0, stream>>>(S, NN, (long)NPOINT * NN,
                                            xyz, NN, (long)3 * NN,
                                            SX, 3, (long)NPOINT * 3,
                                            NPOINT, 3, NN);
    }
    newxyz_kernel<<<BB * 3 * NPOINT / 256, 256, 0, stream>>>(SX, out + OUT_NEWXYZ);

    /* 7. edge assembly: broadcast channels + fused gather-GEMMs (WMMA) */
    edge_base_kernel<<<MCOLS / 256, 256, 0, stream>>>(SP, SX, E);
    {
        dim3 g(NN / 64, NPOINT / 32, BB);   /* sgp - sp -> channels 64..127 */
        wmma_gemm_gather_edge<<<g, 256, 0, stream>>>(S, idx, points, SP, E, DD, 64, NSAMPLE * DD);
        dim3 g2((NSAMPLE * 3 + 63) / 64, NPOINT / 32, BB);  /* sgx - cx -> 131..133 */
        wmma_gemm_gather_edge<<<g2, 256, 0, stream>>>(S, idx, xyz, SX, E, 3, 131, NSAMPLE * 3);
    }

    /* 8. conv stack (WMMA) + BN + leaky relu */
    {
        dim3 g0(MCOLS / 64, 128 / 32, 1);
        wmma_gemm_nn<<<g0, 256, 0, stream>>>(conv_w[0], 134, E, MCOLS, X1, MCOLS, 128, MCOLS, 134);
        bnstats_rows_kernel<<<128, 256, 0, stream>>>(X1, meanb, rstdb);
        bn_apply_leaky_kernel<<<(int)((long)128 * MCOLS / 256), 256, 0, stream>>>(
            X1, 128, bng[0], bnb[0], meanb, rstdb);

        wmma_gemm_nn<<<g0, 256, 0, stream>>>(conv_w[1], 128, X1, MCOLS, X2, MCOLS, 128, MCOLS, 128);
        bnstats_rows_kernel<<<128, 256, 0, stream>>>(X2, meanb, rstdb);
        bn_apply_leaky_kernel<<<(int)((long)128 * MCOLS / 256), 256, 0, stream>>>(
            X2, 128, bng[1], bnb[1], meanb, rstdb);

        dim3 g2(MCOLS / 64, 256 / 32, 1);
        wmma_gemm_nn<<<g2, 256, 0, stream>>>(conv_w[2], 128, X2, MCOLS, X3, MCOLS, 256, MCOLS, 128);
        bnstats_rows_kernel<<<256, 256, 0, stream>>>(X3, meanb, rstdb);
        bn_apply_leaky_kernel<<<(int)((long)256 * MCOLS / 256), 256, 0, stream>>>(
            X3, 256, bng[2], bnb[2], meanb, rstdb);
    }

    /* 9. max over nsample -> new_points */
    maxk_kernel<<<(int)((long)BB * 256 * NPOINT / 256), 256, 0, stream>>>(X3, out + OUT_NEWPTS);
}
